// NextCharLM_13469017440979
// MI455X (gfx1250) — compile-verified, matrix operand order checked by reading, not run on hardware
//
#include <hip/hip_runtime.h>
#include <hip/hip_bf16.h>
#include <math.h>

// ---- model constants (match reference) ----
#define V_     256
#define T_     2048
#define B_     4
#define D_     768
#define L_     6
#define N_     16
#define P_     64
#define DCONV_ 4
#define DI_    1536      // EXPAND*D
#define H_     24        // DI/P
#define DFF_   3072      // 4*D
#define PROJ_  3128      // 2*DI + 2*N + H
#define EPS_   1e-6f
#define M_     (B_*T_)   // 8192 rows

typedef __attribute__((ext_vector_type(16))) __bf16 v16bf;
typedef __attribute__((ext_vector_type(8)))  float  v8f;

__device__ __forceinline__ __bf16 f2bf(float f) {
  union { float f; unsigned u; } in; in.f = f;
  unsigned r = in.u + 0x7fffu + ((in.u >> 16) & 1u);   // round-to-nearest-even
  union { unsigned short s; __bf16 b; } out; out.s = (unsigned short)(r >> 16);
  return out.b;
}

__device__ __forceinline__ float siluf(float x) { return x / (1.f + expf(-x)); }

// CDNA5 async global->LDS copy (ASYNCcnt); builtins don't cover these -> inline asm.
__device__ __forceinline__ void async_ld_b128(unsigned lds_byte_addr, const void* gaddr) {
  asm volatile("global_load_async_to_lds_b128 %0, %1, off"
               :: "v"(lds_byte_addr), "v"(gaddr) : "memory");
}
__device__ __forceinline__ void wait_async4() { asm volatile("s_wait_asynccnt 0x4" ::: "memory"); }
__device__ __forceinline__ void wait_async0() { asm volatile("s_wait_asynccnt 0x0" ::: "memory"); }

// ---------------- embedding + positional ----------------
__global__ void k_embed(const int* __restrict__ idx, const float* __restrict__ emb,
                        const float* __restrict__ pos, float* __restrict__ x) {
  int m = blockIdx.x;
  int tok = idx[m];
  int t = m % T_;
  for (int c = threadIdx.x; c < D_; c += blockDim.x)
    x[(size_t)m * D_ + c] = emb[(size_t)tok * D_ + c] + pos[(size_t)t * D_ + c];
}

// ---------------- RMSNorm (one block per row) -> bf16 out ----------------
__global__ void k_rmsnorm(const float* __restrict__ x, const float* __restrict__ w,
                          __bf16* __restrict__ y) {
  __shared__ float red[8];
  int m = blockIdx.x;
  const float* xr = x + (size_t)m * D_;
  float s = 0.f;
  for (int c = threadIdx.x; c < D_; c += 256) { float v = xr[c]; s += v * v; }
  #pragma unroll
  for (int o = 16; o > 0; o >>= 1) s += __shfl_xor(s, o, 32);
  int lane = threadIdx.x & 31, wid = threadIdx.x >> 5;
  if (lane == 0) red[wid] = s;
  __syncthreads();
  if (threadIdx.x == 0) {
    float t2 = 0.f;
    #pragma unroll
    for (int i = 0; i < 8; i++) t2 += red[i];
    red[0] = rsqrtf(t2 / (float)D_ + EPS_);
  }
  __syncthreads();
  float rs = red[0];
  for (int c = threadIdx.x; c < D_; c += 256)
    y[(size_t)m * D_ + c] = f2bf(xr[c] * rs * w[c]);
}

// ---------------- tiled transpose + cast: out[Nn][K] bf16 <- in[K][Nn] f32 ----------------
__global__ void k_tcast(const float* __restrict__ in, __bf16* __restrict__ out, int K, int Nn) {
  __shared__ float tile[32][33];
  int kb = blockIdx.x * 32, nb = blockIdx.y * 32;
  for (int i = threadIdx.y; i < 32; i += 8) {
    int k = kb + i, n = nb + threadIdx.x;
    tile[i][threadIdx.x] = (k < K && n < Nn) ? in[(size_t)k * Nn + n] : 0.f;
  }
  __syncthreads();
  for (int i = threadIdx.y; i < 32; i += 8) {
    int n = nb + i, k = kb + threadIdx.x;
    if (n < Nn && k < K) out[(size_t)n * K + k] = f2bf(tile[threadIdx.x][i]);
  }
}

// ---------------- plain cast (embed rows already n-major) ----------------
__global__ void k_cast(const float* __restrict__ in, __bf16* __restrict__ out, size_t n) {
  size_t i = (size_t)blockIdx.x * 256 + threadIdx.x;
  if (i < n) out[i] = f2bf(in[i]);
}

// ---------------- causal depthwise conv + SiLU ----------------
__global__ void k_conv(const float* __restrict__ proj, const float* __restrict__ cw,
                       const float* __restrict__ cb, float* __restrict__ u) {
  int c = blockIdx.x * 256 + threadIdx.x;   // channel < DI
  int m = blockIdx.y;
  int t = m % T_; int b = m / T_;
  float acc = cb[c];
  #pragma unroll
  for (int j = 0; j < DCONV_; j++) {
    int tt = t - (DCONV_ - 1) + j;
    if (tt >= 0) acc += proj[(size_t)(b * T_ + tt) * PROJ_ + c] * cw[c * DCONV_ + j];
  }
  u[(size_t)m * DI_ + c] = siluf(acc);
}

// ---------------- B/C normalize + softplus(dt) ----------------
__global__ void k_bcdt(const float* __restrict__ proj, const float* __restrict__ dtbias,
                       float* __restrict__ Bn, float* __restrict__ Cn, float* __restrict__ dtb) {
  int m = blockIdx.x;
  int tid = threadIdx.x;
  const float* pr = proj + (size_t)m * PROJ_;
  if (tid < 32) {
    int n = tid & 15;
    int off = 2 * DI_ + (tid < 16 ? 0 : N_);
    float v = pr[off + n];
    float s = v * v;
    s += __shfl_xor(s, 1, 32); s += __shfl_xor(s, 2, 32);
    s += __shfl_xor(s, 4, 32); s += __shfl_xor(s, 8, 32);
    float out = v * rsqrtf(s + EPS_);
    if (tid < 16) Bn[(size_t)m * N_ + n] = out;
    else          Cn[(size_t)m * N_ + n] = out;
  } else {
    int h = tid - 32;
    if (h < H_) {
      float v = pr[2 * DI_ + 2 * N_ + h] + dtbias[h];
      dtb[(size_t)m * H_ + h] = (v > 20.f) ? v : log1pf(expf(v));
    }
  }
}

// ---------------- SSM scan: one block per (b,h), thread = (p,n) ----------------
__global__ __launch_bounds__(1024) void k_scan(
    const float* __restrict__ u, const float* __restrict__ Bn,
    const float* __restrict__ Cn, const float* __restrict__ dtb,
    const float* __restrict__ A_log, float* __restrict__ out) {
  int bh = blockIdx.x; int b = bh / H_; int h = bh % H_;
  int p = threadIdx.x >> 4, n = threadIdx.x & 15;
  float a = expf(A_log[h * N_ + n]);
  float y = 0.f, z = 0.f;
  size_t base = (size_t)b * T_;
  for (int t = 0; t < T_; t++) {
    size_t m = base + t;
    float ut = u[m * DI_ + h * P_ + p];
    float bt = Bn[m * N_ + n];
    float ct = Cn[m * N_ + n];
    float d  = dtb[m * H_ + h];
    float bu = bt * ut;
    z = (z + d * (bu - a * y)) / (1.f + d * d * a);
    y = y + d * z;
    float o = ct * y;
    o += __shfl_xor(o, 1, 32); o += __shfl_xor(o, 2, 32);
    o += __shfl_xor(o, 4, 32); o += __shfl_xor(o, 8, 32);
    if (n == 0) out[m * DI_ + h * P_ + p] = o;
  }
}

// ---------------- gate: sb = bf16( scan * silu(gate) ) ----------------
__global__ void k_gate(const float* __restrict__ proj, const float* __restrict__ sc,
                       __bf16* __restrict__ sb) {
  size_t i = (size_t)blockIdx.x * 256 + threadIdx.x;
  size_t m = i / DI_; int c = (int)(i % DI_);
  float g = proj[m * PROJ_ + DI_ + c];
  sb[i] = f2bf(sc[i] * siluf(g));
}

// ---------------- bf16 WMMA GEMM with async double-buffered LDS ----------------
// C[M, Nn] = act(A[M,K]bf16 @ B[Nn,K]bf16 (n-major) + bias) + Cadd
// block: 256 threads (8 wave32), tile 128(M) x 128(N), K-step 32, ping-pong LDS via ASYNCcnt.
// Wave = 32(M) x 64(N): 2 A-frags x 4 B-frags -> 8 WMMA per K-step.
template<int ACT, int OUTBF>
__global__ __launch_bounds__(256) void k_gemm(
    const __bf16* __restrict__ A, const __bf16* __restrict__ Bw,
    const float* __restrict__ bias, const float* __restrict__ Cadd,
    void* __restrict__ Cout, int K, int Nn) {
  // row pitch 48 bf16 = 96 B (16B-aligned chunks, staggered banks)
  __shared__ __align__(16) __bf16 As[2][128][48];
  __shared__ __align__(16) __bf16 Bs[2][128][48];
  int tid = threadIdx.x;
  int lane = tid & 31, wid = tid >> 5;
  int wm = wid & 3, wn = wid >> 2;               // 4 waves over M, 2 over N
  int m0 = blockIdx.y * 128;
  int n0 = blockIdx.x * 128;

  v8f acc[2][4];
  #pragma unroll
  for (int i = 0; i < 2; i++)
    #pragma unroll
    for (int j = 0; j < 4; j++) acc[i][j] = (v8f){0.f,0.f,0.f,0.f,0.f,0.f,0.f,0.f};

  // 4 async b128 per thread per tile-pair: 2 A chunks + 2 B chunks
  auto issue_tiles = [&](int buf, int k0) {
    #pragma unroll
    for (int i = 0; i < 2; i++) {          // A: 128 rows x 4 chunks = 512
      int e = tid + i * 256;
      int r = e >> 2, c8 = (e & 3) * 8;
      unsigned lds = (unsigned)(size_t)&As[buf][r][c8];
      async_ld_b128(lds, A + (size_t)(m0 + r) * K + k0 + c8);
    }
    #pragma unroll
    for (int i = 0; i < 2; i++) {          // B: 128 rows x 4 chunks = 512
      int e = tid + i * 256;
      int r = e >> 2, c8 = (e & 3) * 8;
      int grow = n0 + r; if (grow >= Nn) grow = Nn - 1;   // clamp: junk cols never stored
      unsigned lds = (unsigned)(size_t)&Bs[buf][r][c8];
      async_ld_b128(lds, Bw + (size_t)grow * K + k0 + c8);
    }
  };

  int nk = K >> 5;
  issue_tiles(0, 0);
  for (int kt = 0; kt < nk; ++kt) {
    int cur = kt & 1;
    if (kt + 1 < nk) { issue_tiles(cur ^ 1, (kt + 1) << 5); wait_async4(); }
    else             { wait_async0(); }
    __syncthreads();

    // Fragments: lane<16 -> row=lane&15, K {0..7}u{16..23}; lane>=16 -> K {8..15}u{24..31}
    int kd = (lane >> 4) * 4;              // dword offset 0 or 4 within row
    union { v16bf v; unsigned u[8]; } af[2];
    #pragma unroll
    for (int mt = 0; mt < 2; mt++) {
      const unsigned* ap = (const unsigned*)&As[cur][wm * 32 + mt * 16 + (lane & 15)][0];
      #pragma unroll
      for (int j = 0; j < 4; j++) { af[mt].u[j] = ap[kd + j]; af[mt].u[4 + j] = ap[8 + kd + j]; }
    }
    union { v16bf v; unsigned u[8]; } bfv[4];
    #pragma unroll
    for (int nt = 0; nt < 4; nt++) {
      const unsigned* bp = (const unsigned*)&Bs[cur][wn * 64 + nt * 16 + (lane & 15)][0];
      #pragma unroll
      for (int j = 0; j < 4; j++) { bfv[nt].u[j] = bp[kd + j]; bfv[nt].u[4 + j] = bp[8 + kd + j]; }
    }
    #pragma unroll
    for (int mt = 0; mt < 2; mt++)
      #pragma unroll
      for (int nt = 0; nt < 4; nt++)
        acc[mt][nt] = __builtin_amdgcn_wmma_f32_16x16x32_bf16(
            false, af[mt].v, false, bfv[nt].v, (short)0, acc[mt][nt], false, false);

    __syncthreads();   // all waves done reading `cur` before next overwrite
  }

  // store: VGPR r / lane -> row = r + 8*(lane>=16), col = lane&15
  #pragma unroll
  for (int mt = 0; mt < 2; mt++) {
    int rbase = m0 + wm * 32 + mt * 16 + ((lane >> 4) << 3);
    #pragma unroll
    for (int nt = 0; nt < 4; nt++) {
      int col = n0 + wn * 64 + nt * 16 + (lane & 15);
      if (col < Nn) {
        float bv = bias ? bias[col] : 0.f;
        #pragma unroll
        for (int r = 0; r < 8; r++) {
          size_t o = (size_t)(rbase + r) * Nn + col;
          float v = acc[mt][nt][r] + bv;
          if (ACT == 1) {  // gelu (tanh approx)
            float x3 = v * v * v;
            v = 0.5f * v * (1.f + tanhf(0.79788456f * (v + 0.044715f * x3)));
          }
          if (Cadd) v += Cadd[o];
          if (OUTBF) ((__bf16*)Cout)[o] = f2bf(v);
          else       ((float*)Cout)[o]  = v;
        }
      }
    }
  }
}

// ---------------- host orchestration ----------------
extern "C" void kernel_launch(void* const* d_in, const int* in_sizes, int n_in,
                              void* d_out, int out_size, void* d_ws, size_t ws_size,
                              hipStream_t stream) {
  (void)in_sizes; (void)n_in; (void)out_size; (void)ws_size;
  const int*   idx     = (const int*)  d_in[0];
  const float* emb     = (const float*)d_in[1];
  const float* pos     = (const float*)d_in[2];
  const float* rms1    = (const float*)d_in[3];
  const float* rms2    = (const float*)d_in[4];
  const float* rmsf    = (const float*)d_in[5];
  const float* in_w    = (const float*)d_in[6];
  const float* in_b    = (const float*)d_in[7];
  const float* conv_w  = (const float*)d_in[8];
  const float* conv_b  = (const float*)d_in[9];
  const float* A_log   = (const float*)d_in[10];
  const float* dt_bias = (const float*)d_in[11];
  const float* out_w   = (const float*)d_in[12];
  const float* out_b   = (const float*)d_in[13];
  const float* ff1_w   = (const float*)d_in[14];
  const float* ff1_b   = (const float*)d_in[15];
  const float* ff2_w   = (const float*)d_in[16];
  const float* ff2_b   = (const float*)d_in[17];

  const size_t M = (size_t)M_;
  float*  ws     = (float*)d_ws;
  float*  xbuf   = ws;                         // M*D      f32 residual stream
  float*  proj   = xbuf  + M * D_;             // M*PROJ   f32
  float*  ubuf   = proj  + M * PROJ_;          // M*DI     f32 conv out
  float*  scbuf  = ubuf  + M * DI_;            // M*DI     f32 scan out
  float*  Bn     = scbuf + M * DI_;            // M*N
  float*  Cn     = Bn    + M * N_;             // M*N
  float*  dtb    = Cn    + M * N_;             // M*H
  __bf16* hbf    = (__bf16*)(dtb + M * H_);    // M*D      bf16 normed activations
  __bf16* scanbf = hbf    + M * D_;            // M*DI     bf16 gated scan
  __bf16* ffbbf  = scanbf + M * DI_;           // M*DFF    bf16 gelu(ff1)
  __bf16* embbf  = ffbbf  + M * DFF_;          // V*D      bf16 embed (logits B)
  __bf16* wInT   = embbf  + (size_t)V_ * D_;   // [PROJ][D]
  __bf16* wOutT  = wInT   + (size_t)PROJ_ * D_;// [D][DI]
  __bf16* wF1T   = wOutT  + (size_t)D_ * DI_;  // [DFF][D]
  __bf16* wF2T   = wF1T   + (size_t)DFF_ * D_; // [D][DFF]

  dim3 b256(256), bT(32, 8);
  k_embed<<<dim3((unsigned)M), b256, 0, stream>>>(idx, emb, pos, xbuf);
  k_cast <<<dim3((V_ * D_ + 255) / 256), b256, 0, stream>>>(emb, embbf, (size_t)V_ * D_);

  for (int l = 0; l < L_; l++) {
    // per-layer weight conversion (transpose to n-major bf16)
    k_tcast<<<dim3((D_ + 31) / 32, (PROJ_ + 31) / 32), bT, 0, stream>>>(
        in_w + (size_t)l * D_ * PROJ_, wInT, D_, PROJ_);
    k_tcast<<<dim3((DI_ + 31) / 32, (D_ + 31) / 32), bT, 0, stream>>>(
        out_w + (size_t)l * DI_ * D_, wOutT, DI_, D_);
    k_tcast<<<dim3((D_ + 31) / 32, (DFF_ + 31) / 32), bT, 0, stream>>>(
        ff1_w + (size_t)l * D_ * DFF_, wF1T, D_, DFF_);
    k_tcast<<<dim3((DFF_ + 31) / 32, (D_ + 31) / 32), bT, 0, stream>>>(
        ff2_w + (size_t)l * DFF_ * D_, wF2T, DFF_, D_);

    k_rmsnorm<<<dim3((unsigned)M), b256, 0, stream>>>(xbuf, rms1 + (size_t)l * D_, hbf);
    k_gemm<0,0><<<dim3((PROJ_ + 127) / 128, M_ / 128), b256, 0, stream>>>(
        hbf, wInT, in_b + (size_t)l * PROJ_, nullptr, proj, D_, PROJ_);
    k_conv<<<dim3(DI_ / 256, (unsigned)M), b256, 0, stream>>>(
        proj, conv_w + (size_t)l * DI_ * DCONV_, conv_b + (size_t)l * DI_, ubuf);
    k_bcdt<<<dim3((unsigned)M), dim3(64), 0, stream>>>(
        proj, dt_bias + (size_t)l * H_, Bn, Cn, dtb);
    k_scan<<<dim3(B_ * H_), dim3(1024), 0, stream>>>(
        ubuf, Bn, Cn, dtb, A_log + (size_t)l * H_ * N_, scbuf);
    k_gate<<<dim3((unsigned)(M * DI_ / 256)), b256, 0, stream>>>(proj, scbuf, scanbf);
    k_gemm<0,0><<<dim3(D_ / 128, M_ / 128), b256, 0, stream>>>(
        scanbf, wOutT, out_b + (size_t)l * D_, xbuf, xbuf, DI_, D_);
    k_rmsnorm<<<dim3((unsigned)M), b256, 0, stream>>>(xbuf, rms2 + (size_t)l * D_, hbf);
    k_gemm<1,1><<<dim3(DFF_ / 128, M_ / 128), b256, 0, stream>>>(
        hbf, wF1T, ff1_b + (size_t)l * DFF_, nullptr, ffbbf, D_, DFF_);
    k_gemm<0,0><<<dim3(D_ / 128, M_ / 128), b256, 0, stream>>>(
        ffbbf, wF2T, ff2_b + (size_t)l * D_, xbuf, xbuf, DFF_, D_);
  }

  k_rmsnorm<<<dim3((unsigned)M), b256, 0, stream>>>(xbuf, rmsf, hbf);
  // logits = h @ embed^T : embed is already [V][D] = n-major
  k_gemm<0,0><<<dim3((V_ + 127) / 128, M_ / 128), b256, 0, stream>>>(
      hbf, embbf, nullptr, nullptr, (float*)d_out, D_, V_);
}
